// RWKV7_22179211116823
// MI455X (gfx1250) — matrix operand order repaired and checked by reading, not compiled, not run on hardware
//
#include <hip/hip_runtime.h>
#include <hip/hip_bf16.h>

// RWKV7 forward for MI455X (gfx1250, wave32, WMMA).
// Dense matmuls use v_wmma_f32_16x16x32_f16 with f32 accumulate and 2x2
// register tiling per wave (A/B fragment reuse). Weights are transposed +
// converted to f16 [N,K] on device so the WMMA B-fragment (K = e + 16*laneHi,
// order chosen here for A and B) is one contiguous 32B read per lane and the A-fragment is
// two contiguous 16B reads.

typedef _Float16 f16;
typedef __attribute__((ext_vector_type(8)))  f16   v8h;
typedef __attribute__((ext_vector_type(16))) f16   v16h;
typedef __attribute__((ext_vector_type(8)))  float v8f;

#define BB    2
#define TT    512
#define BT    1024          // BB*TT rows
#define DD    768
#define HH    12
#define NN    64
#define NBLK  6
#define VOCAB 8192
#define DRNK  64
#define ARNK  64
#define VRNK  32
#define GRNK  128
#define DFF   1536
#define GN_EPS_C  64e-5f
#define W_SCALE_C (-0.6065306597126334f)

// ---------------------------------------------------------------- transpose
// Wt[n*K + k] = (f16)W[k*N + n]
__global__ __launch_bounds__(256)
void tr_k(const float* __restrict__ W, f16* __restrict__ Wt, int K, int N) {
    int i = blockIdx.x * 256 + threadIdx.x;
    if (i >= K * N) return;
    int k = i / N, n = i % N;
    Wt[(size_t)n * K + k] = (f16)W[i];
}

// ---------------------------------------------------------------- WMMA GEMM
// C[M,N](f32) = A[M,K](f16 row-major) @ Bt[N,K](f16 row-major, pre-transposed)
// Each wave owns a 32x32 output block = 2x2 WMMA tiles (fragment reuse).
// flags: 1 = accumulate into C (residual), 2 = relu. bias/C/C16 nullable.
__global__ __launch_bounds__(256)
void gemm16_f16(const f16* __restrict__ A, const f16* __restrict__ Bt,
                const float* __restrict__ bias, float* __restrict__ C,
                f16* __restrict__ C16, int M, int N, int K, int flags) {
    const int wave = threadIdx.x >> 5;
    const int lane = threadIdx.x & 31;
    const int tm = (blockIdx.y * 2 + (wave >> 2)) * 32;   // block: 64M x 128N
    const int tn = (blockIdx.x * 4 + (wave & 3)) * 32;
    if (tm >= M || tn >= N) return;
    const int l15 = lane & 15;
    const int lh  = lane >> 4;
    const f16* __restrict__ ar0 = A  + (size_t)(tm + l15) * K;
    const f16* __restrict__ ar1 = A  + (size_t)(tm + 16 + l15) * K;
    const f16* __restrict__ br0 = Bt + (size_t)(tn + l15) * K;
    const f16* __restrict__ br1 = Bt + (size_t)(tn + 16 + l15) * K;
    v8f a00 = {}, a01 = {}, a10 = {}, a11 = {};
    for (int k0 = 0; k0 < K; k0 += 32) {
        // A 16-bit 16x32 layout: element e -> K = e + 16*lh (matches B)
        v8h x0 = *(const v8h*)(ar0 + k0 + 16 * lh);
        v8h x1 = *(const v8h*)(ar0 + k0 + 8 + 16 * lh);
        v16h fa0 = __builtin_shufflevector(x0, x1,
                     0,1,2,3,4,5,6,7,8,9,10,11,12,13,14,15);
        v8h y0 = *(const v8h*)(ar1 + k0 + 16 * lh);
        v8h y1 = *(const v8h*)(ar1 + k0 + 8 + 16 * lh);
        v16h fa1 = __builtin_shufflevector(y0, y1,
                     0,1,2,3,4,5,6,7,8,9,10,11,12,13,14,15);
        // B 32x16 layout: element e -> K = e + 16*lh, contiguous in Bt row n
        v16h fb0 = *(const v16h*)(br0 + k0 + 16 * lh);
        v16h fb1 = *(const v16h*)(br1 + k0 + 16 * lh);
        a00 = __builtin_amdgcn_wmma_f32_16x16x32_f16(false, fa0, false, fb0,
                                                     (short)0, a00, false, false);
        a01 = __builtin_amdgcn_wmma_f32_16x16x32_f16(false, fa0, false, fb1,
                                                     (short)0, a01, false, false);
        a10 = __builtin_amdgcn_wmma_f32_16x16x32_f16(false, fa1, false, fb0,
                                                     (short)0, a10, false, false);
        a11 = __builtin_amdgcn_wmma_f32_16x16x32_f16(false, fa1, false, fb1,
                                                     (short)0, a11, false, false);
        // stream-ahead hint (global_prefetch_b8); speculative, drop-safe
        __builtin_prefetch(br0 + k0 + 512, 0, 0);
        __builtin_prefetch(ar0 + k0 + 512, 0, 0);
    }
    auto store = [&](v8f acc, int tmq, int tnq) {
        const int n  = tnq + l15;
        const int mb = tmq + 8 * lh;     // C/D layout: vgpr v -> row mb+v
        const float bv_ = bias ? bias[n] : 0.0f;
#pragma unroll
        for (int i = 0; i < 8; ++i) {
            size_t idx = (size_t)(mb + i) * N + n;
            float r = acc[i] + bv_;
            if (flags & 1) r += C[idx];
            if (flags & 2) r = fmaxf(r, 0.0f);
            if (C)   C[idx]   = r;
            if (C16) C16[idx] = (f16)r;
        }
    };
    store(a00, tm, tn);
    store(a01, tm, tn + 16);
    store(a10, tm + 16, tn);
    store(a11, tm + 16, tn + 16);
}

// ---------------------------------------------------------------- LayerNorm
__global__ __launch_bounds__(256)
void ln_row(const float* __restrict__ in, const float* __restrict__ w,
            const float* __restrict__ b, float* __restrict__ o32,
            f16* __restrict__ o16, int D, float eps) {
    int row = blockIdx.x;
    const float* xp = in + (size_t)row * D;
    float s = 0.f, s2 = 0.f;
    for (int i = threadIdx.x; i < D; i += 256) {
        float v = xp[i]; s += v; s2 += v * v;
    }
    __shared__ float rs[8], rq[8];
    for (int o = 16; o >= 1; o >>= 1) { s += __shfl_xor(s, o); s2 += __shfl_xor(s2, o); }
    int wv = threadIdx.x >> 5;
    if ((threadIdx.x & 31) == 0) { rs[wv] = s; rq[wv] = s2; }
    __syncthreads();
    if (threadIdx.x == 0) {
        float a = 0.f, c = 0.f;
        for (int i = 0; i < 8; ++i) { a += rs[i]; c += rq[i]; }
        rs[0] = a; rq[0] = c;
    }
    __syncthreads();
    float mu  = rs[0] / D;
    float var = rq[0] / D - mu * mu;
    float inv = rsqrtf(var + eps);
    for (int i = threadIdx.x; i < D; i += 256) {
        float v = (xp[i] - mu) * inv * w[i] + b[i];
        if (o32) o32[(size_t)row * D + i] = v;
        if (o16) o16[(size_t)row * D + i] = (f16)v;
    }
}

// ---------------------------------------------------------------- embedding
__global__ __launch_bounds__(256)
void embed_k(const int* __restrict__ audio, const float* __restrict__ emb,
             float* __restrict__ o) {
    int row = blockIdx.x;
    int tok = audio[row];
    for (int i = threadIdx.x; i < DD; i += 256)
        o[(size_t)row * DD + i] = emb[(size_t)tok * DD + i];
}

// ----------------------------------------------------- token-shift mixing x6
__global__ __launch_bounds__(256)
void mix6(const float* __restrict__ h,
          const float* mr, const float* mw, const float* mk,
          const float* mv, const float* ma, const float* mg,
          f16* xr, f16* xw, f16* xk, f16* xv, f16* xa, f16* xg) {
    int row = blockIdx.x;
    int t = row % TT;
    size_t base = (size_t)row * DD;
    for (int i = threadIdx.x; i < DD; i += 256) {
        float cur  = h[base + i];
        float prev = (t == 0) ? 0.f : h[base - DD + i];
        float dx = prev - cur;
        xr[base+i] = (f16)(cur + dx * mr[i]);
        xw[base+i] = (f16)(cur + dx * mw[i]);
        xk[base+i] = (f16)(cur + dx * mk[i]);
        xv[base+i] = (f16)(cur + dx * mv[i]);
        xa[base+i] = (f16)(cur + dx * ma[i]);
        xg[base+i] = (f16)(cur + dx * mg[i]);
    }
}

// ---------------------------------------------------------------- activations
// mode 0: tanh ; 1: sigmoid ; 2: exp(W_SCALE*sigmoid(x))  (decay -> e^w)
__global__ __launch_bounds__(256)
void act_k(const float* __restrict__ in, const float* __restrict__ bias,
           float* __restrict__ o32, f16* __restrict__ o16,
           int cols, int mode, int total) {
    int i = blockIdx.x * 256 + threadIdx.x;
    if (i >= total) return;
    int col = i % cols;
    float v = in[i] + (bias ? bias[col] : 0.f);
    if (mode == 0)      v = tanhf(v);
    else if (mode == 1) v = 1.f / (1.f + expf(-v));
    else                v = expf(W_SCALE_C / (1.f + expf(-v)));
    if (o32) o32[i] = v;
    if (o16) o16[i] = (f16)v;
}

// v = v + (v_first - v) * sigmoid(v0 + t)
__global__ __launch_bounds__(256)
void vmix_k(float* __restrict__ v, const float* __restrict__ vf,
            const float* __restrict__ t, const float* __restrict__ v0) {
    int i = blockIdx.x * 256 + threadIdx.x;
    if (i >= BT * DD) return;
    int col = i % DD;
    float s = 1.f / (1.f + expf(-(v0[col] + t[i])));
    float vv = v[i];
    v[i] = vv + (vf[i] - vv) * s;
}

// per (row,head): kk = l2norm(k*k_k); aneg=-kk; bpos=kk*a; k=k*(1+(a-1)*k_a)
__global__ __launch_bounds__(64)
void kk_k(float* __restrict__ k, const float* __restrict__ a,
          const float* __restrict__ kkv, const float* __restrict__ kav,
          float* __restrict__ aneg, float* __restrict__ bpos) {
    int row = blockIdx.x / HH, h = blockIdx.x % HH;
    int t = threadIdx.x;
    int c = h * NN + t;
    size_t idx = (size_t)row * DD + c;
    float kv  = k[idx];
    float tmp = kv * kkv[c];
    float s = tmp * tmp;
    __shared__ float sh[2];
    for (int o = 16; o >= 1; o >>= 1) s += __shfl_xor(s, o);
    if ((t & 31) == 0) sh[t >> 5] = s;
    __syncthreads();
    float inv = rsqrtf(sh[0] + sh[1] + 1e-12f);
    float kkn = tmp * inv;
    float av  = a[idx];
    aneg[idx] = -kkn;
    bpos[idx] = kkn * av;
    k[idx]    = kv * (1.f + (av - 1.f) * kav[c]);
}

// ------------------------------------------------------------------- scan
// 24 chains (B*H); 256 threads: thread owns row vi (64) x 16-wide k chunk.
__global__ __launch_bounds__(256)
void scan_k(const float* __restrict__ r, const float* __restrict__ wexp,
            const float* __restrict__ k, const float* __restrict__ v,
            const float* __restrict__ an, const float* __restrict__ bp,
            float* __restrict__ y) {
    int b = blockIdx.x / HH, h = blockIdx.x % HH;
    int vi = threadIdx.x >> 2;
    int kq = (threadIdx.x & 3) << 4;
    float S[16];
#pragma unroll
    for (int j = 0; j < 16; ++j) S[j] = 0.f;
    size_t base = (size_t)b * TT * DD + (size_t)h * NN;
    for (int t = 0; t < TT; ++t, base += DD) {
        size_t off = base + kq;
        float al[16], wl[16], kl[16], bl[16], rl[16];
#pragma unroll
        for (int q = 0; q < 4; ++q) {                 // aligned b128 loads
            ((float4*)al)[q] = ((const float4*)(an   + off))[q];
            ((float4*)wl)[q] = ((const float4*)(wexp + off))[q];
            ((float4*)kl)[q] = ((const float4*)(k    + off))[q];
            ((float4*)bl)[q] = ((const float4*)(bp   + off))[q];
            ((float4*)rl)[q] = ((const float4*)(r    + off))[q];
        }
        float vs = v[base + vi];
        float sa = 0.f;
#pragma unroll
        for (int j = 0; j < 16; ++j) sa += S[j] * al[j];
        sa += __shfl_xor(sa, 1); sa += __shfl_xor(sa, 2);  // group-of-4 dot
        float yv = 0.f;
#pragma unroll
        for (int j = 0; j < 16; ++j) {
            S[j] = S[j] * wl[j] + sa * bl[j] + vs * kl[j];
            yv  += S[j] * rl[j];
        }
        yv += __shfl_xor(yv, 1); yv += __shfl_xor(yv, 2);
        if ((threadIdx.x & 3) == 0) y[base + vi] = yv;
    }
}

// ----------------------------- GroupNorm(heads) + r*k*r_k term + output gate
__global__ __launch_bounds__(64)
void gn_gate(const float* __restrict__ y, const float* __restrict__ r,
             const float* __restrict__ k, const float* __restrict__ v,
             const float* __restrict__ g, const float* __restrict__ gnw,
             const float* __restrict__ gnb, const float* __restrict__ rk,
             f16* __restrict__ og16) {
    int row = blockIdx.x / HH, h = blockIdx.x % HH;
    int t = threadIdx.x;
    int c = h * NN + t;
    size_t idx = (size_t)row * DD + c;
    float yv = y[idx];
    float d  = r[idx] * k[idx] * rk[h * NN + t];
    float s = yv, s2 = yv * yv, sd = d;
    __shared__ float sh[6];
    for (int o = 16; o >= 1; o >>= 1) {
        s += __shfl_xor(s, o); s2 += __shfl_xor(s2, o); sd += __shfl_xor(sd, o);
    }
    int wv = t >> 5;
    if ((t & 31) == 0) { sh[wv] = s; sh[2 + wv] = s2; sh[4 + wv] = sd; }
    __syncthreads();
    float S1 = sh[0] + sh[1], S2 = sh[2] + sh[3], SD = sh[4] + sh[5];
    float mu = S1 / NN, var = S2 / NN - mu * mu;
    float o = (yv - mu) * rsqrtf(var + GN_EPS_C) * gnw[c] + gnb[c];
    o += SD * v[idx];
    og16[idx] = (f16)(o * g[idx]);
}

// =========================================================== host launcher
extern "C" void kernel_launch(void* const* d_in, const int* in_sizes, int n_in,
                              void* d_out, int out_size, void* d_ws, size_t ws_size,
                              hipStream_t stream) {
    (void)in_sizes; (void)n_in; (void)out_size; (void)ws_size;
    auto F = [&](int i) { return (const float*)d_in[i]; };
    // Flattening assumption: depth-first dict-insertion order of setup_inputs.
    const int* audio = (const int*)d_in[0];
    const float* emb   = F(1);
    const float* nin_w = F(2), *nin_b = F(3);
    const float* nout_w = F(4), *nout_b = F(5);
    const float* lm_w  = F(6), *lm_b = F(7);

    // ---- workspace bump allocator
    char* wbase = (char*)d_ws;
    size_t woff = 0;
    auto alloc = [&](size_t bytes) -> void* {
        woff = (woff + 255) & ~(size_t)255;
        void* p = wbase + woff;
        woff += bytes;
        return p;
    };
    const size_t BTD = (size_t)BT * DD;
    float* x      = (float*)alloc(BTD * 4);
    float* vfirst = (float*)alloc(BTD * 4);
    float* hbuf   = (float*)alloc(BTD * 4);
    float* rbuf   = (float*)alloc(BTD * 4);
    float* kbuf   = (float*)alloc(BTD * 4);
    float* vbuf   = (float*)alloc(BTD * 4);
    float* abuf   = (float*)alloc(BTD * 4);
    float* wexp   = (float*)alloc(BTD * 4);
    float* gbuf   = (float*)alloc(BTD * 4);
    float* aneg   = (float*)alloc(BTD * 4);
    float* bpos   = (float*)alloc(BTD * 4);
    float* tup    = (float*)alloc(BTD * 4);
    float* ybuf   = (float*)alloc(BTD * 4);
    f16* xr16 = (f16*)alloc(BTD * 2);
    f16* xw16 = (f16*)alloc(BTD * 2);
    f16* xk16 = (f16*)alloc(BTD * 2);
    f16* xv16 = (f16*)alloc(BTD * 2);
    f16* xa16 = (f16*)alloc(BTD * 2);
    f16* xg16 = (f16*)alloc(BTD * 2);
    f16* og16 = (f16*)alloc(BTD * 2);
    f16* h16  = (f16*)alloc(BTD * 2);
    f16* mlph16 = (f16*)alloc((size_t)BT * DFF * 2);
    float* t1  = (float*)alloc((size_t)BT * GRNK * 4);
    f16*   t1h = (f16*)alloc((size_t)BT * GRNK * 2);
    f16* wreg = (f16*)alloc((size_t)DD * VOCAB * 2);   // reused weight region

    auto TR = [&](const float* W, f16* Wt, int K, int N) {
        int tot = K * N;
        tr_k<<<dim3((tot + 255) / 256), dim3(256), 0, stream>>>(W, Wt, K, N);
    };
    auto GEMM = [&](const f16* A, const f16* Bt, const float* bias,
                    float* C, f16* C16, int M, int N, int K, int flags) {
        dim3 g((N + 127) / 128, (M + 63) / 64);
        gemm16_f16<<<g, dim3(256), 0, stream>>>(A, Bt, bias, C, C16, M, N, K, flags);
    };
    auto ACT = [&](const float* in, const float* bias, float* o32, f16* o16,
                   int cols, int mode) {
        int tot = BT * cols;
        act_k<<<dim3((tot + 255) / 256), dim3(256), 0, stream>>>(
            in, bias, o32, o16, cols, mode, tot);
    };

    // embedding + input LN
    embed_k<<<dim3(BT), dim3(256), 0, stream>>>(audio, emb, hbuf);
    ln_row<<<dim3(BT), dim3(256), 0, stream>>>(hbuf, nin_w, nin_b, x,
                                               (f16*)nullptr, DD, 1e-5f);

    // weight-region offsets (in f16 elements)
    const size_t oWr = 0, oWk = 589824, oWv = 1179648, oWo = 1769472;
    const size_t ow1 = 2359296, ow2 = ow1 + 49152, oa1 = ow2 + 49152,
                 oa2 = oa1 + 49152, ov1 = oa2 + 49152, ov2 = ov1 + 24576,
                 og1 = ov2 + 24576, og2 = og1 + 98304, om1 = og2 + 98304,
                 om2 = om1 + 1179648;

    for (int blk = 0; blk < NBLK; ++blk) {
        int pb = 8 + blk * 34;
        const float *ln1w = F(pb+0), *ln1b = F(pb+1), *ln2w = F(pb+2), *ln2b = F(pb+3);
        const float *mxr = F(pb+4), *mxw = F(pb+5), *mxk = F(pb+6);
        const float *mxv = F(pb+7), *mxa = F(pb+8), *mxg = F(pb+9);
        const float *Wr = F(pb+10), *Wk = F(pb+11), *Wv = F(pb+12), *Wo = F(pb+13);
        const float *w0 = F(pb+14), *w1 = F(pb+15), *w2 = F(pb+16);
        const float *a0 = F(pb+17), *a1 = F(pb+18), *a2 = F(pb+19);
        const float *v0 = F(pb+20), *v1 = F(pb+21), *v2 = F(pb+22);
        const float *g1 = F(pb+23), *g2 = F(pb+24);
        const float *k_k = F(pb+25), *k_a = F(pb+26), *r_k = F(pb+27);
        const float *gnw = F(pb+28), *gnb = F(pb+29);
        const float *mW1 = F(pb+30), *mb1 = F(pb+31), *mW2 = F(pb+32), *mb2 = F(pb+33);

        // stage this block's weights as f16 [N,K] (stream order serializes reuse)
        TR(Wr, wreg + oWr, DD, DD);  TR(Wk, wreg + oWk, DD, DD);
        TR(Wv, wreg + oWv, DD, DD);  TR(Wo, wreg + oWo, DD, DD);
        TR(w1, wreg + ow1, DD, DRNK); TR(w2, wreg + ow2, DRNK, DD);
        TR(a1, wreg + oa1, DD, ARNK); TR(a2, wreg + oa2, ARNK, DD);
        if (blk) { TR(v1, wreg + ov1, DD, VRNK); TR(v2, wreg + ov2, VRNK, DD); }
        TR(g1, wreg + og1, DD, GRNK); TR(g2, wreg + og2, GRNK, DD);
        TR(mW1, wreg + om1, DD, DFF); TR(mW2, wreg + om2, DFF, DD);

        // ln1 + token-shift mixes
        ln_row<<<dim3(BT), dim3(256), 0, stream>>>(x, ln1w, ln1b, hbuf,
                                                   (f16*)nullptr, DD, 1e-5f);
        mix6<<<dim3(BT), dim3(256), 0, stream>>>(hbuf, mxr, mxw, mxk, mxv, mxa,
                                                 mxg, xr16, xw16, xk16, xv16,
                                                 xa16, xg16);
        // main projections
        GEMM(xr16, wreg + oWr, nullptr, rbuf, nullptr, BT, DD, DD, 0);
        GEMM(xk16, wreg + oWk, nullptr, kbuf, nullptr, BT, DD, DD, 0);
        GEMM(xv16, wreg + oWv, nullptr, vbuf, nullptr, BT, DD, DD, 0);
        // decay LoRA: wexp = exp(W_SCALE * sigmoid(w0 + tanh(xw@w1)@w2))
        GEMM(xw16, wreg + ow1, nullptr, t1, nullptr, BT, DRNK, DD, 0);
        ACT(t1, nullptr, nullptr, t1h, DRNK, 0);
        GEMM(t1h, wreg + ow2, nullptr, tup, nullptr, BT, DD, DRNK, 0);
        ACT(tup, w0, wexp, nullptr, DD, 2);
        // a LoRA: a = sigmoid(a0 + (xa@a1)@a2)
        GEMM(xa16, wreg + oa1, nullptr, nullptr, t1h, BT, ARNK, DD, 0);
        GEMM(t1h, wreg + oa2, nullptr, tup, nullptr, BT, DD, ARNK, 0);
        ACT(tup, a0, abuf, nullptr, DD, 1);
        // v residual mix
        if (blk == 0) {
            hipMemcpyAsync(vfirst, vbuf, BTD * 4, hipMemcpyDeviceToDevice, stream);
        } else {
            GEMM(xv16, wreg + ov1, nullptr, nullptr, t1h, BT, VRNK, DD, 0);
            GEMM(t1h, wreg + ov2, nullptr, tup, nullptr, BT, DD, VRNK, 0);
            vmix_k<<<dim3((int)((BTD + 255) / 256)), dim3(256), 0, stream>>>(
                vbuf, vfirst, tup, v0);
        }
        // output gate: g = sigmoid(xg@g1) @ g2
        GEMM(xg16, wreg + og1, nullptr, t1, nullptr, BT, GRNK, DD, 0);
        ACT(t1, nullptr, nullptr, t1h, GRNK, 1);
        GEMM(t1h, wreg + og2, nullptr, gbuf, nullptr, BT, DD, GRNK, 0);
        // kk / k update, then the sequential delta-rule scan
        kk_k<<<dim3(BT * HH), dim3(64), 0, stream>>>(kbuf, abuf, k_k, k_a,
                                                     aneg, bpos);
        scan_k<<<dim3(BB * HH), dim3(256), 0, stream>>>(rbuf, wexp, kbuf, vbuf,
                                                        aneg, bpos, ybuf);
        // groupnorm + r*k*r_k*v term + gate, then Wo with residual into x
        gn_gate<<<dim3(BT * HH), dim3(64), 0, stream>>>(ybuf, rbuf, kbuf, vbuf,
                                                        gbuf, gnw, gnb, r_k, og16);
        GEMM(og16, wreg + oWo, nullptr, x, nullptr, BT, DD, DD, 1);
        // MLP
        ln_row<<<dim3(BT), dim3(256), 0, stream>>>(x, ln2w, ln2b,
                                                   (float*)nullptr, h16, DD, 1e-5f);
        GEMM(h16, wreg + om1, mb1, nullptr, mlph16, BT, DFF, DD, 2);
        GEMM(mlph16, wreg + om2, mb2, x, nullptr, BT, DD, DFF, 1);
    }

    // final LN + LM head
    ln_row<<<dim3(BT), dim3(256), 0, stream>>>(x, nout_w, nout_b,
                                               (float*)nullptr, h16, DD, 1e-5f);
    TR(lm_w, wreg, DD, VOCAB);
    GEMM(h16, wreg, lm_b, (float*)d_out, nullptr, BT, VOCAB, DD, 0);
}
